// ScannedRNN_52922587021604
// MI455X (gfx1250) — matrix operand order repaired and checked by reading, not compiled, not run on hardware
//
#include <hip/hip_runtime.h>
#include <hip/hip_bf16.h>
#include <math.h>

// ---------------------------------------------------------------------------
// GRU scan for MI455X (gfx1250): bf16 WMMA (f32 accum) fused step kernel.
// T=512, B=512, H=256.  Recurrent state kept as f32 (epilogue math) + bf16
// shadow (WMMA A operand) so the serial path avoids f32->bf16 conversion.
// All 6 B fragments are loaded before the WMMA chain so ds-load latency is
// overlapped instead of exposed before every WMMA.
// ---------------------------------------------------------------------------

#define T_STEPS 512
#define BATCH   512
#define HID     256
#define HID3    768

typedef float  v8f   __attribute__((ext_vector_type(8)));
typedef __bf16 v16bf __attribute__((ext_vector_type(16)));

static __device__ __forceinline__ __bf16 f2bf(float x) { return (__bf16)x; }

// A-fragment (16x32 bf16) from a row-major f32 source.
// Lanes 0-15 hold row M=lane%16, K = koff+0..7 (elems 0..7) and koff+16..23
// (elems 8..15), koff = (lane<16 ? 0 : 8).
static __device__ __forceinline__ v16bf load_a_from_f32(const float* p) {
  float4 f0 = ((const float4*)p)[0];
  float4 f1 = ((const float4*)p)[1];
  float4 f2 = ((const float4*)(p + 16))[0];
  float4 f3 = ((const float4*)(p + 16))[1];
  v16bf a;
  a[0]  = f2bf(f0.x); a[1]  = f2bf(f0.y); a[2]  = f2bf(f0.z); a[3]  = f2bf(f0.w);
  a[4]  = f2bf(f1.x); a[5]  = f2bf(f1.y); a[6]  = f2bf(f1.z); a[7]  = f2bf(f1.w);
  a[8]  = f2bf(f2.x); a[9]  = f2bf(f2.y); a[10] = f2bf(f2.z); a[11] = f2bf(f2.w);
  a[12] = f2bf(f3.x); a[13] = f2bf(f3.y); a[14] = f2bf(f3.z); a[15] = f2bf(f3.w);
  return a;
}

// A-fragment straight from a row-major bf16 source, with reset-zero select.
static __device__ __forceinline__ v16bf load_a_from_bf16(const __bf16* p, bool zero) {
  union { uint4 q[2]; v16bf v; } u;
  uint4 a = ((const uint4*)p)[0];        // halves koff+0..7
  uint4 b = ((const uint4*)(p + 16))[0]; // halves koff+16..23
  u.q[0].x = zero ? 0u : a.x;  u.q[0].y = zero ? 0u : a.y;
  u.q[0].z = zero ? 0u : a.z;  u.q[0].w = zero ? 0u : a.w;
  u.q[1].x = zero ? 0u : b.x;  u.q[1].y = zero ? 0u : b.y;
  u.q[1].z = zero ? 0u : b.z;  u.q[1].w = zero ? 0u : b.w;
  return u.v;
}

// B-fragment (32x16 bf16) from LDS: lane = column N, K = koffB+0..7 (elems 0..7)
// and koffB+16..23 (elems 8..15), koffB = (lane<16 ? 0 : 8).  p must be 16B aligned.
static __device__ __forceinline__ v16bf load_b_from_lds(const __bf16* p) {
  union { uint4 q[2]; v16bf v; } u;
  u.q[0] = ((const uint4*)p)[0];
  u.q[1] = ((const uint4*)(p + 16))[0];
  return u.v;
}

static __device__ __forceinline__ float sigmoidf(float x) {
  return 1.0f / (1.0f + __expf(-x));
}

// ---------------------------------------------------------------------------
// Weight prep: Wt[g][f] = bf16(W[f][g]) for Wi and Wh (H x 3H -> 3H x H).
// ---------------------------------------------------------------------------
__global__ void gru_prep_weights(const float* __restrict__ Wi,
                                 const float* __restrict__ Wh,
                                 __bf16* __restrict__ Wt_i,
                                 __bf16* __restrict__ Wt_h) {
  int i = blockIdx.x * blockDim.x + threadIdx.x;
  if (i < HID3 * HID) {
    int g = i / HID;
    int f = i % HID;
    Wt_i[i] = f2bf(Wi[f * HID3 + g]);
    Wt_h[i] = f2bf(Wh[f * HID3 + g]);
  }
}

__global__ void gru_init_h(const float* __restrict__ h0,
                           float* __restrict__ hf,
                           __bf16* __restrict__ hb) {
  int i = blockIdx.x * blockDim.x + threadIdx.x;
  if (i < BATCH * HID) {
    float v = h0[i];
    hf[i] = v;
    hb[i] = f2bf(v);
  }
}

// ---------------------------------------------------------------------------
// One GRU time step. Grid: (4 row-groups, 16 col-tiles), 256 threads (8 waves).
// Wave w of row-group rg owns batch rows [ (rg*8+w)*16 , +16 ) and hidden
// columns [ colTile*16, +16 ).  Six 16x16 f32 accumulators: ir,iz,in,hr,hz,hn.
// ---------------------------------------------------------------------------
#define KCHUNK 64
#define SROW   72  // padded LDS row stride in halves (64 data + 8 pad)

__global__ void __launch_bounds__(256)
gru_step(const float* __restrict__ xs,
         const unsigned char* __restrict__ resets,
         const float* __restrict__ bi,
         const float* __restrict__ bhn,
         const __bf16* __restrict__ Wt_i,
         const __bf16* __restrict__ Wt_h,
         const float* __restrict__ hprev_f,
         const __bf16* __restrict__ hprev_b,
         float* __restrict__ hnext_f,
         __bf16* __restrict__ hnext_b,
         float* __restrict__ out,   // [final_h (B*H)] ++ [ys (T*B*H)]
         int t) {
  __shared__ __bf16 smem[192 * SROW];  // 6 gate-col-groups x 16 cols, K-slice of 64

  const int tid   = threadIdx.x;
  const int lane  = tid & 31;
  const int wave  = tid >> 5;
  const int n     = lane & 15;       // column within 16-wide tile
  const bool hi   = lane >= 16;
  const int koffA = hi ? 8 : 0;      // A-fragment K base
  const int koffB = hi ? 8 : 0;      // B-fragment K base

  const int rowTile = blockIdx.x * 8 + wave;      // 0..31
  const int j0      = blockIdx.y * 16;            // hidden col base, 0..240

  const float* xs_t = xs + ((size_t)t * BATCH) * HID;

  // Per-lane A row (M = lane % 16 for both lane halves).
  const int rowA = rowTile * 16 + n;
  const bool rsA = resets[t * BATCH + rowA] != 0;
  const float*  xrow = xs_t + (size_t)rowA * HID;
  const __bf16* hrow = hprev_b + (size_t)rowA * HID;

  const v8f vzero = {0.f, 0.f, 0.f, 0.f, 0.f, 0.f, 0.f, 0.f};
  v8f accI[3] = {vzero, vzero, vzero};  // ir, iz, in
  v8f accH[3] = {vzero, vzero, vzero};  // hr, hz, hn

  for (int kk0 = 0; kk0 < HID; kk0 += KCHUNK) {
    __syncthreads();
    // Cooperative stage of the K-slice of 192 weight columns into LDS.
    // 192 cols x 64 halves = 1536 uint4 chunks; 6 per thread.
    #pragma unroll
    for (int it = 0; it < 6; ++it) {
      int idx   = tid + it * 256;
      int col   = idx >> 3;          // 0..191
      int chunk = idx & 7;           // 8 halves per chunk
      const __bf16* src;
      if (col < 96) {
        int gate = col >> 4, c = col & 15;
        src = Wt_i + ((size_t)(gate * HID + j0 + c)) * HID;
      } else {
        int cc = col - 96;
        int gate = cc >> 4, c = cc & 15;
        src = Wt_h + ((size_t)(gate * HID + j0 + c)) * HID;
      }
      ((uint4*)smem)[col * (SROW / 8) + chunk] = ((const uint4*)(src + kk0))[chunk];
    }
    __syncthreads();

    #pragma unroll
    for (int ks = 0; ks < KCHUNK / 32; ++ks) {
      const int kk = kk0 + ks * 32;
      // Hoist ALL operand fragments first so the 12 ds_load_b128 + 6
      // global_load_b128 issue as clauses and latency is overlapped,
      // then run the 6 WMMAs back-to-back.
      v16bf bI[3], bH[3];
      #pragma unroll
      for (int g = 0; g < 3; ++g) {
        bI[g] = load_b_from_lds(&smem[(g * 16 + n) * SROW + ks * 32 + koffB]);
        bH[g] = load_b_from_lds(&smem[((96 + g * 16) + n) * SROW + ks * 32 + koffB]);
      }
      v16bf ax = load_a_from_f32(xrow + kk + koffA);
      v16bf ah = load_a_from_bf16(hrow + kk + koffA, rsA);
      #pragma unroll
      for (int g = 0; g < 3; ++g) {
        accI[g] = __builtin_amdgcn_wmma_f32_16x16x32_bf16(
            false, ax, false, bI[g], (short)0, accI[g], false, false);
      }
      #pragma unroll
      for (int g = 0; g < 3; ++g) {
        accH[g] = __builtin_amdgcn_wmma_f32_16x16x32_bf16(
            false, ah, false, bH[g], (short)0, accH[g], false, false);
      }
    }
  }

  // ---- Epilogue: gates in f32, write h_next (f32 + bf16 shadow) / ys / final_h ----
  const int jcol = j0 + n;
  const float bir = bi[jcol];
  const float biz = bi[HID + jcol];
  const float bin = bi[2 * HID + jcol];
  const float bh  = bhn[jcol];
  const int rbase = rowTile * 16 + (hi ? 8 : 0);  // C/D layout: row = v + (hi?8:0)

  float* ys = out + (size_t)BATCH * HID + ((size_t)t * BATCH) * HID;

  #pragma unroll
  for (int v = 0; v < 8; ++v) {
    const int row = rbase + v;
    const bool rs = resets[t * BATCH + row] != 0;
    const float hp = rs ? 0.0f : hprev_f[(size_t)row * HID + jcol];
    const float r  = sigmoidf(accI[0][v] + bir + accH[0][v]);
    const float z  = sigmoidf(accI[1][v] + biz + accH[1][v]);
    const float nn = tanhf(accI[2][v] + bin + r * (accH[2][v] + bh));
    const float hnew = (1.0f - z) * nn + z * hp;
    hnext_f[(size_t)row * HID + jcol] = hnew;
    hnext_b[(size_t)row * HID + jcol] = f2bf(hnew);
    ys[(size_t)row * HID + jcol]      = hnew;
    if (t == T_STEPS - 1) out[(size_t)row * HID + jcol] = hnew;  // final_h
  }

  // Warm next step's x rows in L2 (speculative prefetch -> global_prefetch_b8).
  if (t + 1 < T_STEPS) {
    __builtin_prefetch(xs_t + (size_t)BATCH * HID + (size_t)rowA * HID, 0, 1);
  }
}

// ---------------------------------------------------------------------------
// Host launcher.  d_in order: xs, resets, h0, Wi, bi, Wh, bhn.
// d_out: [final_h (B*H)] ++ [ys (T*B*H)], f32.
// ws layout: Wt_i | Wt_h (bf16 768x256 each) | h_f32 ping | h_f32 pong |
//            h_bf16 ping | h_bf16 pong.
// ---------------------------------------------------------------------------
extern "C" void kernel_launch(void* const* d_in, const int* in_sizes, int n_in,
                              void* d_out, int out_size, void* d_ws, size_t ws_size,
                              hipStream_t stream) {
  const float*         xs     = (const float*)d_in[0];
  const unsigned char* resets = (const unsigned char*)d_in[1];
  const float*         h0     = (const float*)d_in[2];
  const float*         Wi     = (const float*)d_in[3];
  const float*         bi     = (const float*)d_in[4];
  const float*         Wh     = (const float*)d_in[5];
  const float*         bhn    = (const float*)d_in[6];
  float*               out    = (float*)d_out;

  char* ws = (char*)d_ws;
  const size_t wbytes = (size_t)HID3 * HID * sizeof(__bf16);   // 384 KB
  const size_t hfloats = (size_t)BATCH * HID;
  __bf16* Wt_i = (__bf16*)ws;
  __bf16* Wt_h = (__bf16*)(ws + wbytes);
  float*  hf0  = (float*)(ws + 2 * wbytes);
  float*  hf1  = hf0 + hfloats;
  __bf16* hb0  = (__bf16*)(hf1 + hfloats);
  __bf16* hb1  = hb0 + hfloats;

  gru_prep_weights<<<(HID3 * HID + 255) / 256, 256, 0, stream>>>(Wi, Wh, Wt_i, Wt_h);
  gru_init_h<<<(BATCH * HID + 255) / 256, 256, 0, stream>>>(h0, hf0, hb0);

  for (int t = 0; t < T_STEPS; ++t) {
    float*  hpf = (t & 1) ? hf1 : hf0;
    float*  hnf = (t & 1) ? hf0 : hf1;
    __bf16* hpb = (t & 1) ? hb1 : hb0;
    __bf16* hnb = (t & 1) ? hb0 : hb1;
    gru_step<<<dim3(4, 16), 256, 0, stream>>>(xs, resets, bi, bhn,
                                              Wt_i, Wt_h, hpf, hpb, hnf, hnb,
                                              out, t);
  }
}
